// HoTLayer_4750233830203
// MI455X (gfx1250) — matrix operand order repaired and checked by reading, not compiled, not run on hardware
//
#include <hip/hip_runtime.h>
#include <hip/hip_bf16.h>
#include <math.h>
#include <stdint.h>

// ---------------------------------------------------------------------------
// Problem constants (reference: B=8, N=1024, D=1024, H=16, K=7, E=4*D)
// ---------------------------------------------------------------------------
constexpr int B_ = 8;
constexpr int N_ = 1024;
constexpr int D_ = 1024;
constexpr int H_ = 16;
constexpr int HD_ = 64;      // head dim
constexpr int E_ = 4096;     // conv expansion
constexpr int KC_ = 7;       // conv kernel
constexpr int M_ = B_ * N_;  // 8192 token rows

// d_out layout (float slots): x_next | ent | route(int64 -> 2 slots each) | g
constexpr int OUT_ENT   = B_ * N_ * D_;          // 8388608
constexpr int OUT_ROUTE = OUT_ENT + B_;          // 8388616 (8 int64 = 16 slots)
constexpr int OUT_G     = OUT_ROUTE + 2 * B_;    // 8388632

// ---------------------------------------------------------------------------
// Types for CDNA5 WMMA
// ---------------------------------------------------------------------------
typedef __bf16 bf16;
typedef __attribute__((ext_vector_type(16))) __bf16 v16bf;
typedef __attribute__((ext_vector_type(8)))  float  v8f;
typedef __attribute__((ext_vector_type(4)))  unsigned int u32x4;

__device__ __forceinline__ void splitf(float x, bf16* hi, bf16* lo) {
  bf16 h = (bf16)x;
  *hi = h;
  *lo = (bf16)(x - (float)h);
}

__device__ __forceinline__ v16bf load_frag2(const bf16* p0, const bf16* p1) {
  v16bf v;
  ((u32x4*)&v)[0] = *(const u32x4*)p0;
  ((u32x4*)&v)[1] = *(const u32x4*)p1;
  return v;
}
__device__ __forceinline__ v16bf load_frag16(const bf16* p) {
  return load_frag2(p, p + 8);
}

// split-bf16 triple product: near-fp32 GEMM on the bf16 matrix pipe
__device__ __forceinline__ v8f wmma3(v16bf ah, v16bf al, v16bf bh, v16bf bl, v8f c) {
  c = __builtin_amdgcn_wmma_f32_16x16x32_bf16(false, ah, false, bh, (short)0, c, false, false);
  c = __builtin_amdgcn_wmma_f32_16x16x32_bf16(false, ah, false, bl, (short)0, c, false, false);
  c = __builtin_amdgcn_wmma_f32_16x16x32_bf16(false, al, false, bh, (short)0, c, false, false);
  return c;
}

// CDNA5 async global -> LDS direct copy (ASYNCcnt tracked), 16B per lane.
// saddr(uniform base) + vaddr(per-lane byte offset) addressing mode.
__device__ __forceinline__ void async_b128(const void* saddr, unsigned voff_bytes,
                                           const void* lds_dst) {
  unsigned l = (unsigned)(size_t)lds_dst;  // flat->LDS: low 32 bits are LDS addr
  asm volatile("global_load_async_to_lds_b128 %0, %1, %2"
               :
               : "v"(l), "v"(voff_bytes), "s"(saddr)
               : "memory");
}
__device__ __forceinline__ void wait_async_le8() {
  asm volatile("s_wait_asynccnt 0x8" ::: "memory");
}
__device__ __forceinline__ void wait_async_0() {
  asm volatile("s_wait_asynccnt 0x0" ::: "memory");
}

__device__ __forceinline__ float gelu_exact(float x) {
  return 0.5f * x * (1.0f + erff(x * 0.70710678118654752f));
}

// block-wide reductions (blockDim.x == 256)
__device__ __forceinline__ float bsum(float v, float* buf) {
  int tid = threadIdx.x;
  __syncthreads();
  buf[tid] = v;
  __syncthreads();
  for (int s = 128; s >= 1; s >>= 1) {
    if (tid < s) buf[tid] += buf[tid + s];
    __syncthreads();
  }
  return buf[0];
}
__device__ __forceinline__ float bmax(float v, float* buf) {
  int tid = threadIdx.x;
  __syncthreads();
  buf[tid] = v;
  __syncthreads();
  for (int s = 128; s >= 1; s >>= 1) {
    if (tid < s) buf[tid] = fmaxf(buf[tid], buf[tid + s]);
    __syncthreads();
  }
  return buf[0];
}

// ---------------------------------------------------------------------------
// K0: zero small accumulators
// ---------------------------------------------------------------------------
__global__ void zero_kernel(float* a, int n) {
  int i = blockIdx.x * blockDim.x + threadIdx.x;
  if (i < n) a[i] = 0.0f;
}

// ---------------------------------------------------------------------------
// weights: f32 -> (bf16 hi, bf16 lo)
// ---------------------------------------------------------------------------
__global__ __launch_bounds__(256) void convert_split_kernel(const float* __restrict__ src,
                                                            bf16* __restrict__ hi,
                                                            bf16* __restrict__ lo, int n) {
  int i = blockIdx.x * 256 + threadIdx.x;
  if (i < n) splitf(src[i], &hi[i], &lo[i]);
}

// ---------------------------------------------------------------------------
// K1: LayerNorm + score stats + per-batch ||x||^2 (all batches)
// ---------------------------------------------------------------------------
__global__ __launch_bounds__(256) void ln_stats_kernel(const float* __restrict__ x,
                                                       const float* __restrict__ gamma,
                                                       const float* __restrict__ beta,
                                                       float* __restrict__ xn,
                                                       bf16* __restrict__ xnh,
                                                       bf16* __restrict__ xnl,
                                                       float* __restrict__ scores,
                                                       float* __restrict__ ssx) {
  __shared__ float buf[256];
  int row = blockIdx.x;
  int b   = row >> 10;
  int tid = threadIdx.x;
  size_t base = (size_t)row * D_;

  float xv[4];
  float s = 0.0f;
  for (int i = 0; i < 4; ++i) { xv[i] = x[base + tid + i * 256]; s += xv[i]; }
  float mean = bsum(s, buf) * (1.0f / D_);

  float cs = 0.0f;
  for (int i = 0; i < 4; ++i) { float c = xv[i] - mean; cs += c * c; }
  float csq = bsum(cs, buf);
  float var = csq * (1.0f / D_);
  float inv = rsqrtf(var + 1e-5f);

  float sn = 0.0f, sn2 = 0.0f;
  for (int i = 0; i < 4; ++i) {
    int d = tid + i * 256;
    float v = (xv[i] - mean) * inv * gamma[d] + beta[d];
    xn[base + d] = v;
    splitf(v, &xnh[base + d], &xnl[base + d]);
    sn += v; sn2 += v * v;
  }
  float sumn  = bsum(sn, buf);
  float sumn2 = bsum(sn2, buf);
  if (tid == 0) {
    float mean2 = sumn * (1.0f / D_);
    float csq2  = sumn2 - (float)D_ * mean2 * mean2;
    if (csq2 < 0.0f) csq2 = 0.0f;
    float std1 = sqrtf(csq2 / (float)(D_ - 1));
    float den  = std1 + 1e-6f;
    scores[row] = (csq2 * (1.0f / D_)) / (den * den);
    atomicAdd(&ssx[b], csq + (float)D_ * mean * mean);  // raw sum of x^2
  }
}

// ---------------------------------------------------------------------------
// K2: softmax-entropy gate, one block per batch
// ---------------------------------------------------------------------------
__global__ __launch_bounds__(256) void gate_kernel(const float* __restrict__ scores,
                                                   const int* __restrict__ prev_route,
                                                   const float* __restrict__ pHlow,
                                                   const float* __restrict__ pHhigh,
                                                   const float* __restrict__ pEmin,
                                                   const float* __restrict__ pEmax,
                                                   float* __restrict__ out,
                                                   int* __restrict__ route) {
  __shared__ float buf[256];
  int b = blockIdx.x, tid = threadIdx.x;
  const float* s = scores + (size_t)b * N_;
  float sv[4];
  float mx = -1e30f;
  for (int i = 0; i < 4; ++i) { sv[i] = s[tid + i * 256]; mx = fmaxf(mx, sv[i]); }
  mx = bmax(mx, buf);
  float ev[4];
  float es = 0.0f;
  for (int i = 0; i < 4; ++i) { ev[i] = expf(sv[i] - mx); es += ev[i]; }
  float S = bsum(es, buf);
  float hs = 0.0f;
  for (int i = 0; i < 4; ++i) { float p = ev[i] / S; hs += p * logf(p + 1e-9f); }
  float Hsum = bsum(hs, buf);
  if (tid == 0) {
    float raw = -Hsum / logf((float)N_);
    float ent = (raw - pEmin[0]) / (pEmax[0] - pEmin[0] + 1e-6f);
    ent = fminf(fmaxf(ent, 0.0f), 1.0f);
    float Hlow = pHlow[0], Hhigh = pHhigh[0];
    const float delta = 0.02f;
    int pr = prev_route[b];
    bool stayA = (pr == 0) && (ent < Hlow + delta);
    bool stayC = (pr == 2) && (ent > Hhigh - delta);
    bool gA = stayA || (ent < Hlow - delta);
    bool gC = stayC || (ent > Hhigh + delta);
    bool gB = !(gA || gC);
    int rt = gC ? 2 : (gB ? 1 : 0);
    out[OUT_ENT + b] = ent;
    ((long long*)(out + OUT_ROUTE))[b] = (long long)rt;
    out[OUT_G + b * 3 + 0] = gA ? 1.0f : 0.0f;
    out[OUT_G + b * 3 + 1] = gB ? 1.0f : 0.0f;
    out[OUT_G + b * 3 + 2] = gC ? 1.0f : 0.0f;
    route[b] = rt;
  }
}

// ---------------------------------------------------------------------------
// K3: depthwise conv (K=7) + GELU -> h1 (bf16 hi/lo), route==1 batches only
// ---------------------------------------------------------------------------
__global__ __launch_bounds__(256) void dwconv_kernel(const float* __restrict__ xn,
                                                     const float* __restrict__ dw,
                                                     bf16* __restrict__ h1h,
                                                     bf16* __restrict__ h1l,
                                                     const int* __restrict__ route) {
  int row = blockIdx.x;
  int b = row >> 10, n = row & (N_ - 1);
  if (route[b] != 1) return;
  int tid = threadIdx.x;
  for (int i = 0; i < 4; ++i) {
    int d = tid + i * 256;
    float acc = 0.0f;
    for (int kk = 0; kk < KC_; ++kk) {
      int nn = n + kk - (KC_ / 2);
      if (nn >= 0 && nn < N_)
        acc += xn[((size_t)b * N_ + nn) * D_ + d] * dw[d * KC_ + kk];
    }
    float g = gelu_exact(acc);
    size_t idx = (size_t)row * D_ + d;
    splitf(g, &h1h[idx], &h1l[idx]);
  }
}

// ---------------------------------------------------------------------------
// K4: tiled split-bf16 WMMA GEMM: out[m][n] = sum_k A[m][k] * W[n][k]
// 256 threads = 8 waves, block tile 128x128, K-step 32, wave tile 32x64.
// Double-buffered LDS (2 x 40KB dynamic) filled by GLOBAL_LOAD_ASYNC_TO_LDS_B128:
// tile k+1 streams in (ASYNCcnt) while tile k feeds the WMMA pipe.
// mode 0: f32 out (+ optional bias)
// mode 1: GELU -> bf16 hi/lo pair (p0)
// mode 2: +bias, scatter q -> p0, k -> p1, v^T -> p2  (QKV epilogue)
// ---------------------------------------------------------------------------
constexpr int LDS_HALF_PER_BUF = 4 * 128 * 40;  // 20480 halfs = 40960 B per buffer

__global__ __launch_bounds__(256) void gemm_bf16x3_kernel(
    const bf16* __restrict__ Ah, const bf16* __restrict__ Al,
    const bf16* __restrict__ Wh, const bf16* __restrict__ Wl,
    const float* __restrict__ bias, float* __restrict__ outF,
    bf16* __restrict__ p0h, bf16* __restrict__ p0l,
    bf16* __restrict__ p1h, bf16* __restrict__ p1l,
    bf16* __restrict__ p2h, bf16* __restrict__ p2l,
    int Ncols, int Kd, int mode, int need, const int* __restrict__ route) {
  const int m0 = blockIdx.y * 128;
  const int n0 = blockIdx.x * 128;
  if (route[m0 >> 10] != need) return;

  extern __shared__ __align__(16) char smem[];
  bf16* lds = (bf16*)smem;

  const int tid = threadIdx.x;
  const int lane = tid & 31;
  const int w = tid >> 5;
  const int wm = w & 3;   // 4 wave rows  -> 32 m each
  const int wn = w >> 2;  // 2 wave cols  -> 64 n each
  const int lh = lane & 15;
  const bool l16 = lane < 16;

  // issue one 128x32 tile (A hi/lo + W hi/lo) into LDS buffer `buf`:
  // 8 async b128 ops per thread -> ASYNCcnt += 8
  auto issue_tile = [&](int buf, int k0) {
    bf16* dAh = lds + buf * LDS_HALF_PER_BUF;
    bf16* dAl = dAh + 128 * 40;
    bf16* dWh = dAl + 128 * 40;
    bf16* dWl = dWh + 128 * 40;
    for (int it2 = 0; it2 < 2; ++it2) {
      int linear = (tid + it2 * 256) * 8;  // 8 halfs (16 B) per op
      int r = linear >> 5, c = linear & 31;
      unsigned ga = (unsigned)(((size_t)(m0 + r) * Kd + k0 + c) * sizeof(bf16));
      unsigned gw = (unsigned)(((size_t)(n0 + r) * Kd + k0 + c) * sizeof(bf16));
      int lo = r * 40 + c;
      async_b128(Ah, ga, dAh + lo);
      async_b128(Al, ga, dAl + lo);
      async_b128(Wh, gw, dWh + lo);
      async_b128(Wl, gw, dWl + lo);
    }
  };

  v8f acc[2][4];
  for (int i = 0; i < 2; ++i)
    for (int j = 0; j < 4; ++j)
      acc[i][j] = (v8f){0.f, 0.f, 0.f, 0.f, 0.f, 0.f, 0.f, 0.f};

  const int nt = Kd >> 5;
  issue_tile(0, 0);
  for (int it = 0; it < nt; ++it) {
    // prefetch next tile into the other buffer, then wait for current tile.
    // async loads complete in order per wave: <=8 outstanding => current done.
    if (it + 1 < nt) {
      issue_tile((it + 1) & 1, (it + 1) << 5);
      wait_async_le8();
    } else {
      wait_async_0();
    }
    __syncthreads();

    bf16* sAh = lds + (it & 1) * LDS_HALF_PER_BUF;
    bf16* sAl = sAh + 128 * 40;
    bf16* sWh = sAl + 128 * 40;
    bf16* sWl = sWh + 128 * 40;

    v16bf afh[2], afl[2];
    for (int i = 0; i < 2; ++i) {
      int r = wm * 32 + i * 16 + lh;
      int c0 = l16 ? 0 : 16;
      afh[i] = load_frag16(sAh + r * 40 + c0);
      afl[i] = load_frag16(sAl + r * 40 + c0);
    }
    v16bf bfh[4], bfl[4];
    for (int j = 0; j < 4; ++j) {
      int c = wn * 64 + j * 16 + lh;
      int kb = l16 ? 0 : 16;
      bfh[j] = load_frag16(sWh + c * 40 + kb);
      bfl[j] = load_frag16(sWl + c * 40 + kb);
    }
    for (int i = 0; i < 2; ++i)
      for (int j = 0; j < 4; ++j)
        acc[i][j] = wmma3(afh[i], afl[i], bfh[j], bfl[j], acc[i][j]);

    __syncthreads();  // all waves done reading buf before it is refilled
  }

  // epilogue: lane = column, 8 rows per accumulator
  const int ro = l16 ? 0 : 8;
  for (int i = 0; i < 2; ++i) {
    int rbase = m0 + wm * 32 + i * 16 + ro;
    for (int j = 0; j < 4; ++j) {
      int col = n0 + wn * 64 + j * 16 + lh;
      float bv = (bias != nullptr) ? bias[col] : 0.0f;
      for (int r = 0; r < 8; ++r) {
        int row = rbase + r;
        float v = acc[i][j][r] + bv;
        if (mode == 0) {
          outF[(size_t)row * Ncols + col] = v;
        } else if (mode == 1) {
          size_t idx = (size_t)row * Ncols + col;
          float g = gelu_exact(v);
          splitf(g, &p0h[idx], &p0l[idx]);
        } else {  // mode 2: qkv scatter
          if (col < D_) {
            size_t idx = (size_t)row * D_ + col;
            splitf(v, &p0h[idx], &p0l[idx]);
          } else if (col < 2 * D_) {
            size_t idx = (size_t)row * D_ + (col - D_);
            splitf(v, &p1h[idx], &p1l[idx]);
          } else {
            int d = col - 2 * D_;
            int bb = row >> 10, n = row & (N_ - 1);
            size_t idx = ((size_t)bb * D_ + d) * N_ + n;  // V transposed [b][d][n]
            splitf(v, &p2h[idx], &p2l[idx]);
          }
        }
      }
    }
  }
}

// ---------------------------------------------------------------------------
// K5: flash attention (route==2). grid = (N/64 qtiles, H, B); 256 thr = 8 waves.
// wave (wi, wj): wi = q subtile (4x16), wj = key/hd half (2x32).
// ---------------------------------------------------------------------------
__global__ __launch_bounds__(256) void attn_kernel(
    const bf16* __restrict__ qh, const bf16* __restrict__ ql,
    const bf16* __restrict__ kh, const bf16* __restrict__ kl,
    const bf16* __restrict__ vth, const bf16* __restrict__ vtl,
    bf16* __restrict__ oh, bf16* __restrict__ ol,
    const int* __restrict__ route) {
  const int b = blockIdx.z;
  if (route[b] != 2) return;
  const int head = blockIdx.y;
  const int qt = blockIdx.x;

  __shared__ float sS[64][65];
  __shared__ bf16 sPh[64][72];
  __shared__ bf16 sPl[64][72];
  __shared__ float sRed[64][4];
  __shared__ float sSum[64][4];
  __shared__ float sM[64], sLs[64], sFac[64];

  const int tid = threadIdx.x;
  const int lane = tid & 31;
  const int w = tid >> 5;
  const int wi = w & 3;
  const int wj = w >> 2;
  const int lh = lane & 15;
  const bool l16 = lane < 16;

  if (tid < 64) { sM[tid] = -1e30f; sLs[tid] = 0.0f; }
  __syncthreads();

  // Q fragments (persist across the key loop): row = token, K-dim = head features
  const int mrow = qt * 64 + wi * 16 + lh;
  const size_t qbase = ((size_t)b * N_ + mrow) * D_ + head * HD_;
  v16bf qfh[2], qfl[2];
  for (int ks = 0; ks < 2; ++ks) {
    int c0 = ks * 32 + (l16 ? 0 : 16);
    qfh[ks] = load_frag16(qh + qbase + c0);
    qfl[ks] = load_frag16(ql + qbase + c0);
  }

  v8f oacc[2];
  oacc[0] = (v8f){0.f, 0.f, 0.f, 0.f, 0.f, 0.f, 0.f, 0.f};
  oacc[1] = oacc[0];

  const int rr = tid >> 2;   // softmax row (0..63)
  const int seg = tid & 3;   // 16-col segment

  for (int kc = 0; kc < N_ / 64; ++kc) {
    // --- S = (Q K^T) * 1/sqrt(hd), 16x32 per wave ---
    v8f sacc[2];
    sacc[0] = (v8f){0.f, 0.f, 0.f, 0.f, 0.f, 0.f, 0.f, 0.f};
    sacc[1] = sacc[0];
    for (int s = 0; s < 2; ++s) {
      int kt = kc * 64 + wj * 32 + s * 16 + lh;
      const size_t kb = ((size_t)b * N_ + kt) * D_ + head * HD_;
      for (int ks = 0; ks < 2; ++ks) {
        int cb = ks * 32 + (l16 ? 0 : 16);
        v16bf bh = load_frag16(kh + kb + cb);
        v16bf bl = load_frag16(kl + kb + cb);
        sacc[s] = wmma3(qfh[ks], qfl[ks], bh, bl, sacc[s]);
      }
    }
    for (int s = 0; s < 2; ++s) {
      int col = wj * 32 + s * 16 + lh;
      int rb = wi * 16 + (l16 ? 0 : 8);
      for (int r = 0; r < 8; ++r) sS[rb + r][col] = sacc[s][r] * 0.125f;
    }
    __syncthreads();

    // --- online softmax stats ---
    {
      float mx = -1e30f;
      for (int c = seg * 16; c < seg * 16 + 16; ++c) mx = fmaxf(mx, sS[rr][c]);
      sRed[rr][seg] = mx;
    }
    __syncthreads();
    if (tid < 64) {
      float m4 = fmaxf(fmaxf(sRed[tid][0], sRed[tid][1]), fmaxf(sRed[tid][2], sRed[tid][3]));
      float mnew = fmaxf(sM[tid], m4);
      sFac[tid] = expf(sM[tid] - mnew);
      sM[tid] = mnew;
    }
    __syncthreads();
    {
      float mn = sM[rr];
      float ssum = 0.0f;
      for (int c = seg * 16; c < seg * 16 + 16; ++c) {
        float p = expf(sS[rr][c] - mn);
        ssum += p;
        splitf(p, &sPh[rr][c], &sPl[rr][c]);
      }
      sSum[rr][seg] = ssum;
    }
    __syncthreads();
    if (tid < 64) {
      sLs[tid] = sLs[tid] * sFac[tid] +
                 (sSum[tid][0] + sSum[tid][1] + sSum[tid][2] + sSum[tid][3]);
    }

    // --- rescale running O, then O += P * V ---
    {
      int rb = wi * 16 + (l16 ? 0 : 8);
      for (int r = 0; r < 8; ++r) {
        float f = sFac[rb + r];
        oacc[0][r] *= f;
        oacc[1][r] *= f;
      }
    }
    for (int s = 0; s < 2; ++s) {
      int dcol = wj * 32 + s * 16 + lh;
      const size_t vb = ((size_t)b * D_ + head * HD_ + dcol) * N_ + kc * 64;
      for (int ks = 0; ks < 2; ++ks) {
        int prow = wi * 16 + lh;
        int c0 = ks * 32 + (l16 ? 0 : 16);
        v16bf ah = load_frag16(&sPh[prow][c0]);
        v16bf al = load_frag16(&sPl[prow][c0]);
        int cb = ks * 32 + (l16 ? 0 : 16);
        v16bf bh = load_frag16(vth + vb + cb);
        v16bf bl = load_frag16(vtl + vb + cb);
        oacc[s] = wmma3(ah, al, bh, bl, oacc[s]);
      }
    }
    __syncthreads();
  }

  // --- finalize: O / l, write o as bf16 hi/lo [M][D] ---
  for (int s = 0; s < 2; ++s) {
    int dcol = wj * 32 + s * 16 + lh;
    int f = head * HD_ + dcol;
    int rb = wi * 16 + (l16 ? 0 : 8);
    for (int r = 0; r < 8; ++r) {
      int rloc = rb + r;
      float val = oacc[s][r] / sLs[rloc];
      int tok = qt * 64 + rloc;
      size_t idx = ((size_t)b * N_ + tok) * D_ + f;
      splitf(val, &oh[idx], &ol[idx]);
    }
  }
}

// ---------------------------------------------------------------------------
// K6: LayerNorm of conv-path output in place + per-batch ||outB||^2 (route==1)
// ---------------------------------------------------------------------------
__global__ __launch_bounds__(256) void outb_ln_kernel(float* __restrict__ P,
                                                      const float* __restrict__ g,
                                                      const float* __restrict__ bt,
                                                      const int* __restrict__ route,
                                                      float* __restrict__ ssb) {
  int row = blockIdx.x;
  int b = row >> 10;
  if (route[b] != 1) return;
  __shared__ float buf[256];
  int tid = threadIdx.x;
  size_t base = (size_t)row * D_;
  float v[4];
  float s = 0.0f;
  for (int i = 0; i < 4; ++i) { v[i] = P[base + tid + i * 256]; s += v[i]; }
  float mean = bsum(s, buf) * (1.0f / D_);
  float cs = 0.0f;
  for (int i = 0; i < 4; ++i) { float c = v[i] - mean; cs += c * c; }
  float var = bsum(cs, buf) * (1.0f / D_);
  float inv = rsqrtf(var + 1e-5f);
  float s2 = 0.0f;
  for (int i = 0; i < 4; ++i) {
    int d = tid + i * 256;
    float o = (v[i] - mean) * inv * g[d] + bt[d];
    P[base + d] = o;
    s2 += o * o;
  }
  float t = bsum(s2, buf);
  if (tid == 0) atomicAdd(&ssb[b], t);
}

// ---------------------------------------------------------------------------
// K7: masked merge + residual -> d_out
// ---------------------------------------------------------------------------
__global__ __launch_bounds__(256) void merge_kernel(const float* __restrict__ x,
                                                    const float* __restrict__ P,
                                                    const float* __restrict__ alpha,
                                                    const int* __restrict__ route,
                                                    const float* __restrict__ ssx,
                                                    const float* __restrict__ ssb,
                                                    float* __restrict__ out) {
  int row = blockIdx.x;
  int b = row >> 10;
  int tid = threadIdx.x;
  int rt = route[b];
  float a = alpha[0];
  float scale = 1.0f;
  if (rt == 1) scale = sqrtf(ssx[b]) / (sqrtf(ssb[b]) + 1e-6f);
  size_t base = (size_t)row * D_;
  for (int i = 0; i < 4; ++i) {
    size_t idx = base + tid + i * 256;
    float xv = x[idx];
    float path;
    if (rt == 0) path = xv;
    else if (rt == 1) path = P[idx] * scale;
    else path = P[idx];
    out[idx] = xv + a * path;
  }
}

// ---------------------------------------------------------------------------
// Host: workspace layout + launch sequence (all on `stream`, graph-capturable)
// ---------------------------------------------------------------------------
extern "C" void kernel_launch(void* const* d_in, const int* in_sizes, int n_in,
                              void* d_out, int out_size, void* d_ws, size_t ws_size,
                              hipStream_t stream) {
  const float* x          = (const float*)d_in[0];
  const int*   prev_route = (const int*)d_in[1];
  const float* pHlow      = (const float*)d_in[2];
  const float* pHhigh     = (const float*)d_in[3];
  const float* pEmin      = (const float*)d_in[4];
  const float* pEmax      = (const float*)d_in[5];
  const float* gamma_pre  = (const float*)d_in[6];
  const float* beta_pre   = (const float*)d_in[7];
  const float* dw_w       = (const float*)d_in[8];
  const float* expand_w   = (const float*)d_in[9];
  const float* proj_w     = (const float*)d_in[10];
  const float* conv_gamma = (const float*)d_in[11];
  const float* conv_beta  = (const float*)d_in[12];
  const float* in_proj_w  = (const float*)d_in[13];
  const float* in_proj_b  = (const float*)d_in[14];
  const float* out_w      = (const float*)d_in[15];
  const float* out_b      = (const float*)d_in[16];
  const float* alpha      = (const float*)d_in[17];

  float* out = (float*)d_out;

  char* w = (char*)d_ws;
  size_t off = 0;
  auto take = [&](size_t bytes) -> char* {
    char* p = w + off;
    off = (off + bytes + 255) & ~(size_t)255;
    return p;
  };
  const size_t MD = (size_t)M_ * D_;  // 8M elements

  float* xn    = (float*)take(MD * 4);
  bf16*  xnh   = (bf16*)take(MD * 2);
  bf16*  xnl   = (bf16*)take(MD * 2);
  float* scor  = (float*)take((size_t)M_ * 4);
  int*   route = (int*)take(B_ * 4);
  float* ss    = (float*)take(16 * 4);   // [0..7]=||x||^2, [8..15]=||outB||^2
  float* ssx   = ss;
  float* ssb   = ss + 8;
  bf16*  h1h   = (bf16*)take(MD * 2);
  bf16*  h1l   = (bf16*)take(MD * 2);
  bf16*  h2h   = (bf16*)take((size_t)M_ * E_ * 2);
  bf16*  h2l   = (bf16*)take((size_t)M_ * E_ * 2);
  float* P     = (float*)take(MD * 4);   // conv proj out / outC (disjoint batches)
  bf16*  qh    = (bf16*)take(MD * 2);
  bf16*  qlb   = (bf16*)take(MD * 2);
  bf16*  kh    = (bf16*)take(MD * 2);
  bf16*  klb   = (bf16*)take(MD * 2);
  bf16*  vth   = (bf16*)take(MD * 2);
  bf16*  vtl   = (bf16*)take(MD * 2);
  bf16*  ohb   = (bf16*)take(MD * 2);
  bf16*  olb   = (bf16*)take(MD * 2);
  bf16*  ewh   = (bf16*)take((size_t)E_ * D_ * 2);
  bf16*  ewl   = (bf16*)take((size_t)E_ * D_ * 2);
  bf16*  pwh   = (bf16*)take((size_t)D_ * E_ * 2);
  bf16*  pwl   = (bf16*)take((size_t)D_ * E_ * 2);
  bf16*  iwh   = (bf16*)take((size_t)3 * D_ * D_ * 2);
  bf16*  iwl   = (bf16*)take((size_t)3 * D_ * D_ * 2);
  bf16*  owh   = (bf16*)take((size_t)D_ * D_ * 2);
  bf16*  owl   = (bf16*)take((size_t)D_ * D_ * 2);

  // allow 80KB dynamic LDS for the double-buffered GEMM (WGP has 320KB)
  const int GEMM_LDS = 2 * LDS_HALF_PER_BUF * (int)sizeof(bf16);  // 81920 B
  (void)hipFuncSetAttribute((const void*)gemm_bf16x3_kernel,
                            hipFuncAttributeMaxDynamicSharedMemorySize, GEMM_LDS);

  // 0) zero the small accumulators
  zero_kernel<<<1, 32, 0, stream>>>(ss, 16);

  // 1) weight conversion to split-bf16
  {
    int nE = E_ * D_, nI = 3 * D_ * D_, nO = D_ * D_;
    convert_split_kernel<<<(nE + 255) / 256, 256, 0, stream>>>(expand_w, ewh, ewl, nE);
    convert_split_kernel<<<(nE + 255) / 256, 256, 0, stream>>>(proj_w, pwh, pwl, nE);
    convert_split_kernel<<<(nI + 255) / 256, 256, 0, stream>>>(in_proj_w, iwh, iwl, nI);
    convert_split_kernel<<<(nO + 255) / 256, 256, 0, stream>>>(out_w, owh, owl, nO);
  }

  // 2) LN + gate statistics (all batches)
  ln_stats_kernel<<<M_, 256, 0, stream>>>(x, gamma_pre, beta_pre, xn, xnh, xnl, scor, ssx);
  gate_kernel<<<B_, 256, 0, stream>>>(scor, prev_route, pHlow, pHhigh, pEmin, pEmax, out, route);

  // 3) Path B: depthwise conv -> expand (GELU) -> proj -> LN
  dwconv_kernel<<<M_, 256, 0, stream>>>(xn, dw_w, h1h, h1l, route);
  gemm_bf16x3_kernel<<<dim3(E_ / 128, M_ / 128), 256, GEMM_LDS, stream>>>(
      h1h, h1l, ewh, ewl, nullptr, nullptr,
      h2h, h2l, nullptr, nullptr, nullptr, nullptr,
      E_, D_, /*mode=*/1, /*need=*/1, route);
  gemm_bf16x3_kernel<<<dim3(D_ / 128, M_ / 128), 256, GEMM_LDS, stream>>>(
      h2h, h2l, pwh, pwl, nullptr, P,
      nullptr, nullptr, nullptr, nullptr, nullptr, nullptr,
      D_, E_, /*mode=*/0, /*need=*/1, route);
  outb_ln_kernel<<<M_, 256, 0, stream>>>(P, conv_gamma, conv_beta, route, ssb);

  // 4) Path C: QKV -> flash attention -> out projection
  gemm_bf16x3_kernel<<<dim3(3 * D_ / 128, M_ / 128), 256, GEMM_LDS, stream>>>(
      xnh, xnl, iwh, iwl, in_proj_b, nullptr,
      qh, qlb, kh, klb, vth, vtl,
      3 * D_, D_, /*mode=*/2, /*need=*/2, route);
  attn_kernel<<<dim3(N_ / 64, H_, B_), 256, 0, stream>>>(
      qh, qlb, kh, klb, vth, vtl, ohb, olb, route);
  gemm_bf16x3_kernel<<<dim3(D_ / 128, M_ / 128), 256, GEMM_LDS, stream>>>(
      ohb, olb, owh, owl, out_b, P,
      nullptr, nullptr, nullptr, nullptr, nullptr, nullptr,
      D_, D_, /*mode=*/0, /*need=*/2, route);

  // 5) merge + residual
  merge_kernel<<<M_, 256, 0, stream>>>(x, P, alpha, route, ssx, ssb, out);
}
